// GrCNetAttentionLayer_7962869367669
// MI455X (gfx1250) — matrix operand order repaired and checked by reading, compiled, not run on hardware
//
#include <hip/hip_runtime.h>
#include <hip/hip_bf16.h>
#include <cstdint>

typedef __attribute__((ext_vector_type(16))) _Float16 v16h;
typedef __attribute__((ext_vector_type(8)))  float    v8f;
typedef __attribute__((ext_vector_type(2)))  _Float16 h2;

#define LRELU_ALPHA 0.2f

// ---------------------------------------------------------------------------
// Kernel 1: per-node projection via WMMA.
//   U  = X @ A1^T   (f32, [N,64])
//   Vh = X @ A2^T   (f16, [N,64])
//   P  = X @ (A1^T a2), Q = X @ (A2^T a2)   (f32 scalars per node)
// One wave handles a 16-row tile; 8 waves (256 threads) per block.
// ---------------------------------------------------------------------------
__global__ void node_project_wmma(const float* __restrict__ X,    // [N,64]
                                  const float* __restrict__ Aw,   // [64,128]
                                  const float* __restrict__ a2,   // [64]
                                  float* __restrict__ U,
                                  _Float16* __restrict__ Vh,
                                  float* __restrict__ P,
                                  float* __restrict__ Q,
                                  int N)
{
    __shared__ float w1s[64];   // A1^T a2
    __shared__ float w2s[64];   // A2^T a2
    int tid = threadIdx.x;
    if (tid < 128) {
        int k = tid & 63;
        int m = tid >> 6;
        float acc = 0.f;
        #pragma unroll 8
        for (int o = 0; o < 64; ++o)
            acc += a2[o] * Aw[o * 128 + m * 64 + k];
        if (m == 0) w1s[k] = acc; else w2s[k] = acc;
    }
    __syncthreads();

    int wave = tid >> 5;
    int lane = tid & 31;
    int tile = blockIdx.x * (blockDim.x >> 5) + wave;
    int row0 = tile * 16;
    if (row0 >= N) return;              // wave-uniform exit, EXEC stays all-1s

    const bool fullTile = (row0 + 16 <= N);

    int r    = lane & 15;
    int half = lane >> 4;
    int row  = row0 + r;
    int rowc = (row < N) ? row : (N - 1);

    // A fragments: 16x32 f16 tile of X per the CDNA5 16-bit A layout.
    // VGPR g, pos: K = (g<4?0:16) + half*8 + (g&3)*2 + pos  (+ 32*kb)
    v16h afrag[2];
    float pPart = 0.f, qPart = 0.f;
    #pragma unroll
    for (int kb = 0; kb < 2; ++kb) {
        #pragma unroll
        for (int e = 0; e < 16; ++e) {
            int g = e >> 1, pos = e & 1;
            int K = kb * 32 + ((g < 4) ? 0 : 16) + half * 8 + (g & 3) * 2 + pos;
            float xv = X[(size_t)rowc * 64 + K];
            afrag[kb][e] = (_Float16)xv;
            pPart += xv * w1s[K];
            qPart += xv * w2s[K];
        }
    }
    // row r is split across lanes r (K-half 0) and r+16 (K-half 1)
    float pFull = pPart + __shfl_xor(pPart, 16, 32);
    float qFull = qPart + __shfl_xor(qPart, 16, 32);
    if (half == 0 && row < N) { P[row] = pFull; Q[row] = qFull; }

    // m = 0 -> A1 (U accumulator, f32 out); m = 1 -> A2 (V, stored f16)
    #pragma unroll
    for (int m = 0; m < 2; ++m) {
        #pragma unroll
        for (int j = 0; j < 4; ++j) {       // 16-wide output column block
            v8f acc = {};
            #pragma unroll
            for (int kb = 0; kb < 2; ++kb) {
                // B fragment 32x16: element e -> K = (e<8?0:16) + half*8 + (e&7), same as A
                // B[k, o] = Aw[o, m*64 + k], o = j*16 + r
                v16h bfrag;
                #pragma unroll
                for (int e = 0; e < 16; ++e) {
                    int K = kb * 32 + ((e < 8) ? 0 : 16) + half * 8 + (e & 7);
                    bfrag[e] = (_Float16)Aw[(size_t)(j * 16 + r) * 128 + m * 64 + K];
                }
                acc = __builtin_amdgcn_wmma_f32_16x16x32_f16(
                        false, afrag[kb], false, bfrag, (short)0, acc, false, false);
            }
            // D layout: lane -> N=r; VGPR i -> M = i + 8*half
            int o = j * 16 + r;
            size_t base = (size_t)(row0 + 8 * half) * 64 + o;
            if (fullTile) {
                // branch-free fast path (tiles exactly cover N when N % 16 == 0)
                if (m == 0) {
                    #pragma unroll
                    for (int i = 0; i < 8; ++i)
                        U[base + (size_t)i * 64] = acc[i];
                } else {
                    #pragma unroll
                    for (int i = 0; i < 8; ++i)
                        Vh[base + (size_t)i * 64] = (_Float16)acc[i];
                }
            } else {
                #pragma unroll
                for (int i = 0; i < 8; ++i) {
                    int rr = row0 + i + 8 * half;
                    if (rr < N) {
                        if (m == 0) U[(size_t)rr * 64 + o]  = acc[i];
                        else        Vh[(size_t)rr * 64 + o] = (_Float16)acc[i];
                    }
                }
            }
        }
    }
}

// ---------------------------------------------------------------------------
// Kernel 2: one wave per edge. Scalar attention + scatter-add of w * v[dst]
// into h[src]; scalar segment sums into rowsum/Wsum.
// ---------------------------------------------------------------------------
__global__ void edge_scatter(const int* __restrict__ src,
                             const int* __restrict__ dst,
                             const int* __restrict__ etype,
                             const float* __restrict__ G,     // [N,R]
                             const float* __restrict__ P,
                             const float* __restrict__ Q,
                             const _Float16* __restrict__ Vh, // [N,64] f16
                             float* __restrict__ rowsum,
                             float* __restrict__ Wsum,
                             float* __restrict__ H,           // [N,64] f32
                             int E, int R)
{
    int e    = (int)((blockIdx.x * (size_t)blockDim.x + threadIdx.x) >> 5);
    int lane = threadIdx.x & 31;
    if (e >= E) return;

    int s = src[e], d = dst[e], t = etype[e];
    // uniform per-wave scalar math (same-address loads broadcast in hw)
    float gran = 0.5f * (G[(size_t)s * R + t] + G[(size_t)d * R + t]);
    float raw  = gran * (P[s] + Q[d]);
    float lr   = (raw >= 0.f) ? raw : LRELU_ALPHA * raw;
    float ee   = __expf(-lr);
    float w    = ee * gran;

    if (lane == 0) {
        atomicAdd(&rowsum[s], ee);
        atomicAdd(&Wsum[s], w);
    }
    // coalesced 128B gather of v[d] as packed f16 pairs
    h2 hv = ((const h2*)(Vh + (size_t)d * 64))[lane];
    float* hrow = H + (size_t)s * 64;
    atomicAdd(&hrow[2 * lane],     w * (float)hv.x);
    atomicAdd(&hrow[2 * lane + 1], w * (float)hv.y);
}

// ---------------------------------------------------------------------------
// Kernel 3: out = elu( (Wsum[n]*U[n,:] + H[n,:]) / rowsum[n] )
// ---------------------------------------------------------------------------
__global__ void finalize_elu(const float* __restrict__ U,
                             const float* __restrict__ H,
                             const float* __restrict__ rowsum,
                             const float* __restrict__ Wsum,
                             float* __restrict__ out,
                             int N)
{
    int idx = blockIdx.x * blockDim.x + threadIdx.x;
    if (idx >= N * 64) return;
    int n = idx >> 6;
    float rs = rowsum[n];
    rs = (rs == 0.f) ? 1e-12f : rs;
    float v = (Wsum[n] * U[idx] + H[idx]) / rs;
    out[idx] = (v > 0.f) ? v : (__expf(v) - 1.f);
}

extern "C" void kernel_launch(void* const* d_in, const int* in_sizes, int n_in,
                              void* d_out, int out_size, void* d_ws, size_t ws_size,
                              hipStream_t stream)
{
    const float* X     = (const float*)d_in[0];   // [N,64]
    const int*   edge  = (const int*)  d_in[1];   // [2,E]
    //            d_in[2] edge_embed: unused by the reference (204 MB never read)
    const int*   etype = (const int*)  d_in[3];   // [E]
    const float* G     = (const float*)d_in[4];   // [N,R]
    const float* Aw    = (const float*)d_in[5];   // [64,128]
    const float* a2    = (const float*)d_in[6];   // [1,64]
    float* out = (float*)d_out;

    const int N = in_sizes[0] / 64;
    const int E = in_sizes[3];
    const int R = in_sizes[4] / N;
    const int* src = edge;
    const int* dst = edge + E;

    // workspace layout (256B-aligned slices)
    char* ws = (char*)d_ws;
    size_t off = 0;
    auto take = [&](size_t bytes) {
        char* p = ws + off;
        off += (bytes + 255) & ~(size_t)255;
        return p;
    };
    float*    U      = (float*)   take((size_t)N * 64 * sizeof(float));
    _Float16* Vh     = (_Float16*)take((size_t)N * 64 * sizeof(_Float16));
    float*    P      = (float*)   take((size_t)N * sizeof(float));
    float*    Q      = (float*)   take((size_t)N * sizeof(float));
    float*    rowsum = (float*)   take((size_t)N * sizeof(float));
    float*    Wsum   = (float*)   take((size_t)N * sizeof(float));
    float*    H      = (float*)   take((size_t)N * 64 * sizeof(float));
    (void)ws_size; (void)n_in; (void)out_size;

    // zero the accumulators (graph-capturable)
    hipMemsetAsync(rowsum, 0, (size_t)N * sizeof(float), stream);
    hipMemsetAsync(Wsum,   0, (size_t)N * sizeof(float), stream);
    hipMemsetAsync(H,      0, (size_t)N * 64 * sizeof(float), stream);

    // 1) node projection (WMMA): 8 wave-tiles (128 rows) per 256-thread block
    int tiles   = (N + 15) / 16;
    int blocks1 = (tiles + 7) / 8;
    node_project_wmma<<<blocks1, 256, 0, stream>>>(X, Aw, a2, U, Vh, P, Q, N);

    // 2) edge scatter: one wave per edge
    long long threads2 = (long long)E * 32;
    int blocks2 = (int)((threads2 + 255) / 256);
    edge_scatter<<<blocks2, 256, 0, stream>>>(src, dst, etype, G, P, Q, Vh,
                                              rowsum, Wsum, H, E, R);

    // 3) finalize
    int tot = N * 64;
    finalize_elu<<<(tot + 255) / 256, 256, 0, stream>>>(U, H, rowsum, Wsum, out, N);
}
